// GatedJunction_18081812316805
// MI455X (gfx1250) — compile-verified, matrix operand order checked by reading, not run on hardware
//
#include <hip/hip_runtime.h>
#include <hip/hip_bf16.h>

typedef __attribute__((ext_vector_type(16))) _Float16 v16h;
typedef __attribute__((ext_vector_type(8)))  float    v8f;

#define NB   32      // batch
#define NC   256     // channels
#define NK   4       // branches
#define NHW  3136    // 56*56
#define NHW4 784     // NHW/4
#define CIN  1280    // (K+1)*C
#define NHID 32

// ---------------------------------------------------------------------------
// Kernel 1: global average pool. One 256-thread block per (src,b,c) plane.
// src 0 = y, src 1..4 = x[k]. Writes means laid out as w_in[b][src*C + c].
// ---------------------------------------------------------------------------
__global__ void __launch_bounds__(256) pool_kernel(
    const float* __restrict__ y, const float* __restrict__ x,
    float* __restrict__ means)
{
    const int bid = blockIdx.x;
    const int src = bid / (NB * NC);
    const int rem = bid % (NB * NC);      // b*C + c
    const int b   = rem / NC;
    const int c   = rem % NC;

    const float* plane = (src == 0)
        ? (y + (size_t)rem * NHW)
        : (x + ((size_t)(src - 1) * NB * NC + rem) * NHW);
    const float4* p4 = (const float4*)plane;

    float s = 0.f;
    for (int i = threadIdx.x; i < NHW4; i += 256) {
        float4 v = p4[i];
        s += v.x + v.y + v.z + v.w;
    }
    // wave32 reduction
    #pragma unroll
    for (int off = 16; off > 0; off >>= 1) s += __shfl_down(s, off, 32);

    __shared__ float wsum[8];
    const int lane = threadIdx.x & 31;
    const int wv   = threadIdx.x >> 5;
    if (lane == 0) wsum[wv] = s;
    __syncthreads();
    if (wv == 0) {
        s = (lane < 8) ? wsum[lane] : 0.f;
        #pragma unroll
        for (int off = 4; off > 0; off >>= 1) s += __shfl_down(s, off, 32);
        if (lane == 0) means[b * CIN + src * NC + c] = s * (1.f / (float)NHW);
    }
}

// ---------------------------------------------------------------------------
// Kernel 2: the gating MLP on WMMA. One block, 4 waves (wave32).
//  GEMM1: h[32x32] = relu((means @ W1^T) * gamma + beta)   (K = 1280)
//  GEMM2: wraw[32x1280] = h @ W2^T + b2                    (K = 32)
// Fragment layouts per CDNA5 ISA 7.12.2 (16-bit A/B, 32-bit C/D).
// ---------------------------------------------------------------------------
__global__ void __launch_bounds__(128) mlp_wmma_kernel(
    const float* __restrict__ means,   // [32][1280]
    const float* __restrict__ W1,      // [32][1280]
    const float* __restrict__ gamma_,  // [32]
    const float* __restrict__ beta_,   // [32]
    const float* __restrict__ W2,      // [1280][32]
    const float* __restrict__ b2,      // [1280]
    float* __restrict__ wraw)          // [32][1280]
{
    __shared__ _Float16 hlds[32][40];  // padded to dodge bank conflicts

    const int tid  = threadIdx.x;
    const int wave = tid >> 5;
    const int lane = tid & 31;
    const int half = lane >> 4;        // lane group (K-swizzle select)
    const int nl   = lane & 15;        // row (A) / col (B) within tile

    // ----- GEMM1: each wave owns one 16x16 tile of h (2x2 tiling of 32x32)
    {
        const int mi = wave >> 1;
        const int ni = wave & 1;
        v8f c = {};
        for (int kk = 0; kk < CIN; kk += 32) {
            v16h a, bm;
            #pragma unroll
            for (int i = 0; i < 8; ++i) {
                // 16-bit A layout: VGPR i holds K = kb, kb+1
                const int kb = ((i < 4) ? 0 : 16) + (i & 3) * 2 + half * 8 + kk;
                a[2*i]    = (_Float16)means[(mi*16 + nl) * CIN + kb];
                a[2*i+1]  = (_Float16)means[(mi*16 + nl) * CIN + kb + 1];
                // B[k][n] = W1[n][k]  (W1 is [HID][CIN] row-major)
                bm[2*i]   = (_Float16)W1[(ni*16 + nl) * CIN + kb];
                bm[2*i+1] = (_Float16)W1[(ni*16 + nl) * CIN + kb + 1];
            }
            c = __builtin_amdgcn_wmma_f32_16x16x32_f16(
                    false, a, false, bm, (short)0, c, false, false);
        }
        // epilogue: BN (running stats are identity) + ReLU, stash h as f16
        const int   n  = ni*16 + nl;
        const float gm = gamma_[n];
        const float bt = beta_[n];
        #pragma unroll
        for (int r = 0; r < 8; ++r) {
            const int m = mi*16 + r + half*8;   // C/D layout: M = r + 8*half
            float hv = c[r] * gm + bt;
            hlds[m][n] = (_Float16)(hv > 0.f ? hv : 0.f);
        }
    }
    __syncthreads();

    // ----- GEMM2: 2 x 80 tiles of 16x16, one WMMA (K=32) each
    for (int t = wave; t < 160; t += 4) {
        const int mi = t / 80;
        const int nt = t % 80;
        v16h a, bm;
        #pragma unroll
        for (int i = 0; i < 8; ++i) {
            const int kb = ((i < 4) ? 0 : 16) + (i & 3) * 2 + half * 8;
            a[2*i]    = hlds[mi*16 + nl][kb];
            a[2*i+1]  = hlds[mi*16 + nl][kb + 1];
            // B[k][n] = W2[n][k]  (W2 is [CIN][HID] row-major)
            bm[2*i]   = (_Float16)W2[(nt*16 + nl) * NHID + kb];
            bm[2*i+1] = (_Float16)W2[(nt*16 + nl) * NHID + kb + 1];
        }
        v8f c = {};
        c = __builtin_amdgcn_wmma_f32_16x16x32_f16(
                false, a, false, bm, (short)0, c, false, false);
        const int   col = nt*16 + nl;
        const float bb  = b2[col];
        #pragma unroll
        for (int r = 0; r < 8; ++r) {
            const int m = mi*16 + r + half*8;
            wraw[m * CIN + col] = c[r] + bb;
        }
    }
}

// ---------------------------------------------------------------------------
// Kernel 3: gates. w1 = sigmoid(w[:,0,:]); g = softmax over branches.
// ---------------------------------------------------------------------------
__global__ void __launch_bounds__(256) gates_kernel(
    const float* __restrict__ wraw,  // [32][1280]
    float* __restrict__ w1g,         // [B][C]
    float* __restrict__ gsm)         // [B][K][C]
{
    const int idx = blockIdx.x * blockDim.x + threadIdx.x;
    if (idx >= NB * NC) return;
    const int b = idx / NC;
    const int c = idx % NC;
    const float* row = wraw + (size_t)b * CIN;

    w1g[idx] = 1.f / (1.f + __expf(-row[c]));

    float e[NK];
    float mx = -3.0e38f;
    #pragma unroll
    for (int k = 0; k < NK; ++k) {
        e[k] = row[(k + 1) * NC + c];
        mx = fmaxf(mx, e[k]);
    }
    float s = 0.f;
    #pragma unroll
    for (int k = 0; k < NK; ++k) { e[k] = __expf(e[k] - mx); s += e[k]; }
    const float inv = 1.f / s;
    #pragma unroll
    for (int k = 0; k < NK; ++k)
        gsm[((size_t)b * NK + k) * NC + c] = e[k] * inv;
}

// ---------------------------------------------------------------------------
// Kernel 4: out = y*w1 + sum_k g_k * x_k. One block per (b,c) plane,
// float4 streaming — the HBM-bound bulk of the op.
// ---------------------------------------------------------------------------
__global__ void __launch_bounds__(256) amplify_kernel(
    const float* __restrict__ y, const float* __restrict__ x,
    const float* __restrict__ w1g, const float* __restrict__ gsm,
    float* __restrict__ out)
{
    const int bc = blockIdx.x;          // b*C + c
    const int b  = bc / NC;
    const int c  = bc % NC;

    const float wy = w1g[bc];
    float gk[NK];
    #pragma unroll
    for (int k = 0; k < NK; ++k) gk[k] = gsm[((size_t)b * NK + k) * NC + c];

    const float4* y4 = (const float4*)y + (size_t)bc * NHW4;
    const float4* x4 = (const float4*)x;
    float4*       o4 = (float4*)out + (size_t)bc * NHW4;
    const size_t planeStride = (size_t)NB * NC * NHW4;  // float4 units per branch

    for (int i = threadIdx.x; i < NHW4; i += 256) {
        float4 yv = y4[i];
        float4 acc;
        acc.x = yv.x * wy; acc.y = yv.y * wy;
        acc.z = yv.z * wy; acc.w = yv.w * wy;
        #pragma unroll
        for (int k = 0; k < NK; ++k) {
            float4 xv = x4[(size_t)k * planeStride + (size_t)bc * NHW4 + i];
            acc.x += gk[k] * xv.x; acc.y += gk[k] * xv.y;
            acc.z += gk[k] * xv.z; acc.w += gk[k] * xv.w;
        }
        o4[i] = acc;
    }
}

// ---------------------------------------------------------------------------
extern "C" void kernel_launch(void* const* d_in, const int* in_sizes, int n_in,
                              void* d_out, int out_size, void* d_ws, size_t ws_size,
                              hipStream_t stream) {
    const float* y     = (const float*)d_in[0];
    const float* x     = (const float*)d_in[1];
    const float* W1    = (const float*)d_in[2];
    const float* gamma = (const float*)d_in[3];
    const float* beta  = (const float*)d_in[4];
    const float* W2    = (const float*)d_in[5];
    const float* b2    = (const float*)d_in[6];
    float* out = (float*)d_out;

    float* ws    = (float*)d_ws;
    float* means = ws;                     // 40960 f32
    float* wraw  = ws + 40960;             // 40960 f32
    float* w1g   = ws + 81920;             //  8192 f32
    float* gsm   = ws + 90112;             // 32768 f32

    pool_kernel<<<(NK + 1) * NB * NC, 256, 0, stream>>>(y, x, means);
    mlp_wmma_kernel<<<1, 128, 0, stream>>>(means, W1, gamma, beta, W2, b2, wraw);
    gates_kernel<<<(NB * NC + 255) / 256, 256, 0, stream>>>(wraw, w1g, gsm);
    amplify_kernel<<<NB * NC, 256, 0, stream>>>(y, x, w1g, gsm, out);
}
